// MahalanobisDetector_20143396618299
// MI455X (gfx1250) — matrix operand order repaired and checked by reading, not compiled, not run on hardware
//
#include <hip/hip_runtime.h>

// ---------------------------------------------------------------------------
// Mahalanobis OOD detector, MI455X (gfx1250, wave32, WMMA).
// Big GEMMs (x@W, zfP@W^T, sign@W) run as bf16 WMMA with f32 accumulation:
// W (616 MB fp32) streams from HBM twice (pass1: x@W; pass2: fused
// backward+re-forward) -> ~1.2 GB -> ~53us at 23.3 TB/s, with bf16 matrix
// compute below that floor. All reductions fixed-order (deterministic).
// ---------------------------------------------------------------------------

#define B_   128
#define L_   4
#define C_   1000
#define D_   256
#define PIN_ 150528          // 3*224*224
#define HW_  50176           // 224*224
#define MAG_ 1e-3f

#define KC_  3072            // split-K chunk over P
#define NCH_ 49              // PIN_/KC_

// k1 tiling
#define KSTEP_ 64
#define AP_    88            // LDS pitches (x2B multiple of 16B, conflict-friendly)
#define BP_    88

// k3 tiling
#define SUB_ 32              // W rows per sub-step
#define W1P_ 264             // W sub row-major pitch (d)
#define W2P_ 40              // W sub col-major pitch (p)
#define SP_  40              // sign-matrix pitch (p)

typedef __bf16 bf16;
typedef __attribute__((ext_vector_type(16))) __bf16        v16bf;
typedef __attribute__((ext_vector_type(8)))  float         v8f;
typedef __attribute__((ext_vector_type(4)))  unsigned int  u32x4;
typedef __attribute__((ext_vector_type(2)))  unsigned int  u32x2;
typedef __attribute__((ext_vector_type(4)))  float         f32x4;

union Frag { u32x4 q[2]; v16bf v; };

__device__ __forceinline__ unsigned short f2bf(float f) {
  unsigned int u = __float_as_uint(f);
  u += 0x7fffu + ((u >> 16) & 1u);           // round-to-nearest-even
  return (unsigned short)(u >> 16);
}

__device__ __forceinline__ v8f wmma_bf16(v16bf a, v16bf b, v8f c) {
  // (neg_a, A, neg_b, B, c_mod, C, reuse_a, reuse_b)
  return __builtin_amdgcn_wmma_f32_16x16x32_bf16(false, a, false, b, (short)0, c,
                                                 false, false);
}

// ---------------------------------------------------------------------------
// k1: out_part[l][kc][b][d_half] = x[b, chunk] @ W_l[chunk, d_half]   (bf16 WMMA)
// grid = L*NCH*2 halves, block = 512 (16 waves). Wave w: rows mi=w&7 (16 rows),
// cols (w>>3)*64 .. +64 (4 tiles of 16).
// ---------------------------------------------------------------------------
__global__ __launch_bounds__(512) void k1_feat(const float* __restrict__ x,
                                               const float* __restrict__ W,
                                               float* __restrict__ part) {
  __shared__ unsigned short sA[128 * AP_];   // x tile, row-major [b][k]
  __shared__ unsigned short sB[128 * BP_];   // W tile, transposed [n][k]

  const int blk = blockIdx.x;
  const int nh = blk & 1;
  const int kc = (blk >> 1) % NCH_;
  const int l  = blk / (2 * NCH_);
  const int kbase = kc * KC_;
  const int ncol0 = nh * 128;
  const float* Wl = W + (size_t)l * PIN_ * D_;

  const int t = threadIdx.x;
  const int lane = t & 31, wave = t >> 5;
  const int lmod = lane & 15, lhalf = lane >> 4;
  const int mi = wave & 7, nset = wave >> 3;
  const int m0 = mi * 16;

  v8f acc[4] = {};

  for (int kk = 0; kk < KC_; kk += KSTEP_) {
    // stage x (128 x 64) -> bf16 row-major
#pragma unroll
    for (int i = 0; i < 4; ++i) {
      int idx4 = t + i * 512;
      int row = idx4 >> 4;
      int kq  = (idx4 & 15) << 2;
      f32x4 v = *(const f32x4*)(x + (size_t)row * PIN_ + kbase + kk + kq);
      u32x2 pk;
      pk.x = (unsigned)f2bf(v.x) | ((unsigned)f2bf(v.y) << 16);
      pk.y = (unsigned)f2bf(v.z) | ((unsigned)f2bf(v.w) << 16);
      *(u32x2*)(sA + row * AP_ + kq) = pk;
    }
    // stage W (64 x 128) -> bf16 transposed [n][k]; W is streamed (non-temporal)
#pragma unroll
    for (int i = 0; i < 4; ++i) {
      int idx4 = t + i * 512;
      int kr = idx4 >> 5;
      int nq = (idx4 & 31) << 2;
      const float* gp = Wl + (size_t)(kbase + kk + kr) * D_ + ncol0 + nq;
      f32x4 v = __builtin_nontemporal_load((const f32x4*)gp);
      sB[(nq + 0) * BP_ + kr] = f2bf(v.x);
      sB[(nq + 1) * BP_ + kr] = f2bf(v.y);
      sB[(nq + 2) * BP_ + kr] = f2bf(v.z);
      sB[(nq + 3) * BP_ + kr] = f2bf(v.w);
      if (kk + KSTEP_ < KC_) __builtin_prefetch(gp + (size_t)KSTEP_ * D_, 0, 0);
    }
    __syncthreads();

#pragma unroll
    for (int ks = 0; ks < KSTEP_; ks += 32) {
      Frag fa;  // A 16x32 bf16 frag: lanes 0-15 k=0..15, 16-31 k=16..31
      const unsigned short* pa = sA + (m0 + lmod) * AP_ + ks + lhalf * 16;
      fa.q[0] = *(const u32x4*)pa;
      fa.q[1] = *(const u32x4*)(pa + 8);
#pragma unroll
      for (int j = 0; j < 4; ++j) {
        Frag fb;  // B 32x16 frag: lanes 0-15 k=0..15, 16-31 k=16..31
        const unsigned short* pb =
            sB + ((nset * 4 + j) * 16 + lmod) * BP_ + ks + lhalf * 16;
        fb.q[0] = *(const u32x4*)pb;
        fb.q[1] = *(const u32x4*)(pb + 8);
        acc[j] = wmma_bf16(fa.v, fb.v, acc[j]);
      }
    }
    __syncthreads();
  }

  float* pout = part + (size_t)(l * NCH_ + kc) * (B_ * D_) + ncol0;
#pragma unroll
  for (int j = 0; j < 4; ++j) {
    int n = (nset * 4 + j) * 16 + lmod;
#pragma unroll
    for (int r = 0; r < 8; ++r) {
      int m = m0 + r + 8 * lhalf;                 // C/D layout: vgpr r -> row r / r+8
      pout[(size_t)m * D_ + n] = acc[j][r];
    }
  }
}

// ---------------------------------------------------------------------------
// k3: fused backward + re-forward over one W chunk.
//   gx[b,p]  = sum_d zfP[b,d] * W[p,d]          (GEMM1, K=256)
//   s[b,p]   = (gx>=0 ? +1 : -1) / std[chan(p)]
//   corr[b,d]+= sum_p s[b,p] * W[p,d]           (GEMM2, accumulated in VGPRs)
// One pass over W serves both GEMMs. grid = L*NCH, block = 512.
// ---------------------------------------------------------------------------
__global__ __launch_bounds__(512) void k3_fused(const float* __restrict__ W,
                                                const unsigned short* __restrict__ zfPbf,
                                                const float* __restrict__ stdv,
                                                float* __restrict__ part) {
  __shared__ unsigned short sW1[SUB_ * W1P_];  // W sub row-major [p][d] (B of GEMM1)
  __shared__ unsigned short sW2[D_ * W2P_];    // W sub col-major [d][p] (B of GEMM2)
  __shared__ unsigned short sS [B_ * SP_];     // sign matrix [b][p]     (A of GEMM2)

  const int blk = blockIdx.x;
  const int kc = blk % NCH_;
  const int l  = blk / NCH_;
  const int kbase = kc * KC_;
  const float* Wl = W + (size_t)l * PIN_ * D_;
  const unsigned short* zl = zfPbf + (size_t)l * B_ * D_;

  const int t = threadIdx.x;
  const int lane = t & 31, wave = t >> 5;
  const int lmod = lane & 15, lhalf = lane >> 4;
  const int mi = wave & 7;       // b strip
  const int hi = wave >> 3;      // 0/1
  const int m0 = mi * 16;

  const float inv0 = 1.f / stdv[0], inv1 = 1.f / stdv[1], inv2 = 1.f / stdv[2];

  v8f corr[8] = {};

  for (int sub = 0; sub < KC_; sub += SUB_) {
    __syncthreads();  // previous GEMM2 done with sW*/sS

    // stage W rows [kbase+sub, +32) x 256, in both orientations (streamed)
#pragma unroll
    for (int i = 0; i < 4; ++i) {
      int idx4 = t + i * 512;
      int kp = idx4 >> 6;
      int dq = (idx4 & 63) << 2;
      const float* gp = Wl + (size_t)(kbase + sub + kp) * D_ + dq;
      f32x4 v = __builtin_nontemporal_load((const f32x4*)gp);
      unsigned short b0 = f2bf(v.x), b1 = f2bf(v.y), b2 = f2bf(v.z), b3 = f2bf(v.w);
      u32x2 pk;
      pk.x = (unsigned)b0 | ((unsigned)b1 << 16);
      pk.y = (unsigned)b2 | ((unsigned)b3 << 16);
      *(u32x2*)(sW1 + kp * W1P_ + dq) = pk;
      sW2[(dq + 0) * W2P_ + kp] = b0;
      sW2[(dq + 1) * W2P_ + kp] = b1;
      sW2[(dq + 2) * W2P_ + kp] = b2;
      sW2[(dq + 3) * W2P_ + kp] = b3;
      if (sub + SUB_ < KC_) __builtin_prefetch(gp + (size_t)SUB_ * D_, 0, 0);
    }
    __syncthreads();

    // GEMM1: one 16x16 gx tile per wave (b strip mi, p half hi), K = D = 256
    {
      v8f a1 = {};
      const unsigned short* za = zl + (size_t)(m0 + lmod) * D_ + lhalf * 16;
#pragma unroll
      for (int ks = 0; ks < D_; ks += 32) {
        Frag fa;  // zfP from global bf16 (L0-hot, 64KB/layer)
        fa.q[0] = *(const u32x4*)(za + ks);
        fa.q[1] = *(const u32x4*)(za + ks + 8);
        Frag fb;  // B[k=d][n=p] = W[p][d] -> contiguous d in sW1 row p
        const unsigned short* pb = sW1 + (hi * 16 + lmod) * W1P_ + ks + lhalf * 16;
        fb.q[0] = *(const u32x4*)pb;
        fb.q[1] = *(const u32x4*)(pb + 8);
        a1 = wmma_bf16(fa.v, fb.v, a1);
      }
      // sign(gx)/std[channel] -> bf16 s, stored row-major for A-frag reads
      const int pl = hi * 16 + lmod;          // p within sub tile (per-lane const)
      const int pg = kbase + sub + pl;        // global p index
      const float inv = (pg >= 2 * HW_) ? inv2 : (pg >= HW_ ? inv1 : inv0);
      const unsigned short sp = f2bf(inv), sn = f2bf(-inv);
#pragma unroll
      for (int r = 0; r < 8; ++r) {
        int b = m0 + r + 8 * lhalf;
        sS[b * SP_ + pl] = (a1[r] >= 0.f) ? sp : sn;
      }
    }
    __syncthreads();

    // GEMM2: corr[b][d] += s[b][p] * W[p][d], K = 32 (one wmma step)
    {
      Frag fa;
      const unsigned short* pa = sS + (m0 + lmod) * SP_ + lhalf * 16;
      fa.q[0] = *(const u32x4*)pa;
      fa.q[1] = *(const u32x4*)(pa + 8);
#pragma unroll
      for (int j = 0; j < 8; ++j) {
        Frag fb;  // B[k=p][n=d] = W[p][d] -> contiguous p in sW2 row d
        const unsigned short* pb =
            sW2 + ((hi * 8 + j) * 16 + lmod) * W2P_ + lhalf * 16;
        fb.q[0] = *(const u32x4*)pb;
        fb.q[1] = *(const u32x4*)(pb + 8);
        corr[j] = wmma_bf16(fa.v, fb.v, corr[j]);
      }
    }
  }

  float* pout = part + (size_t)(l * NCH_ + kc) * (B_ * D_);
#pragma unroll
  for (int j = 0; j < 8; ++j) {
    int n = (hi * 8 + j) * 16 + lmod;
#pragma unroll
    for (int r = 0; r < 8; ++r) {
      int m = m0 + r + 8 * lhalf;
      pout[(size_t)m * D_ + n] = corr[j][r];
    }
  }
}

// ---------------------------------------------------------------------------
// vP = vin @ P (per layer), vPv = rowwise vP.vin. grid = L*nrows, block = 256.
// ---------------------------------------------------------------------------
__global__ void k_quadprep(const float* __restrict__ vin, const float* __restrict__ Pr,
                           int nrows, float* __restrict__ vP, float* __restrict__ vPv) {
  const int lb = blockIdx.x;
  const int l = lb / nrows;
  const int d = threadIdx.x;
  __shared__ float sv[D_];
  __shared__ float red[D_];
  sv[d] = vin[(size_t)lb * D_ + d];
  __syncthreads();
  const float* Pl = Pr + (size_t)l * D_ * D_;
  float acc = 0.f;
#pragma unroll 4
  for (int e = 0; e < D_; ++e) acc = fmaf(sv[e], Pl[(size_t)e * D_ + d], acc);
  vP[(size_t)lb * D_ + d] = acc;
  red[d] = acc * sv[d];
  __syncthreads();
  for (int s = D_ / 2; s > 0; s >>= 1) {
    if (d < s) red[d] += red[d + s];
    __syncthreads();
  }
  if (d == 0) vPv[lb] = red[0];
}

// ---------------------------------------------------------------------------
// quad scores vs all classes: mode 0 -> argmax (first index on ties), write
// zfP = oP - mP[pred] as bf16; mode 1 -> write max score. grid = L*B.
// ---------------------------------------------------------------------------
__global__ void k_classify(const float* __restrict__ vP, const float* __restrict__ vPv,
                           const float* __restrict__ mu, const float* __restrict__ mP,
                           const float* __restrict__ mPm, int mode,
                           unsigned short* __restrict__ zfPbf,
                           float* __restrict__ scores) {
  const int lb = blockIdx.x;
  const int l = lb / B_;
  const int t = threadIdx.x;  // 256
  __shared__ float so[D_];
  so[t] = vP[(size_t)lb * D_ + t];
  __syncthreads();
  const float xx = vPv[lb];
  float best = -3.0e38f;
  int bidx = C_;
  for (int c = t; c < C_; c += D_) {
    const float* mrow = mu + ((size_t)l * C_ + c) * D_;
    float dot = 0.f;
#pragma unroll 4
    for (int e = 0; e < D_; ++e) dot = fmaf(so[e], mrow[e], dot);
    float g = -0.5f * (xx - 2.f * dot + mPm[l * C_ + c]);
    if (g > best) { best = g; bidx = c; }
  }
  __shared__ float rv[D_];
  __shared__ int   ri[D_];
  rv[t] = best; ri[t] = bidx;
  __syncthreads();
  for (int s = D_ / 2; s > 0; s >>= 1) {
    if (t < s) {
      if (rv[t + s] > rv[t] || (rv[t + s] == rv[t] && ri[t + s] < ri[t])) {
        rv[t] = rv[t + s]; ri[t] = ri[t + s];
      }
    }
    __syncthreads();
  }
  if (mode == 0) {
    int pred = ri[0];
    float z = so[t] - mP[((size_t)l * C_ + pred) * D_ + t];  // (out-mu[pred])@P
    zfPbf[(size_t)lb * D_ + t] = f2bf(z);
  } else if (t == 0) {
    scores[lb] = rv[0];
  }
}

// dst = (base?base:0) + scale * sum_kc part  (fixed-order, deterministic)
__global__ void k_reduce(const float* __restrict__ part, const float* __restrict__ base,
                         float scale, float* __restrict__ dst) {
  const int i = blockIdx.x * blockDim.x + threadIdx.x;
  const int l = i / (B_ * D_);
  const int r = i - l * (B_ * D_);
  float s = 0.f;
  for (int kc = 0; kc < NCH_; ++kc)
    s += part[(size_t)(l * NCH_ + kc) * (B_ * D_) + r];
  dst[i] = (base ? base[i] : 0.f) + scale * s;
}

__global__ void k_final(const float* __restrict__ scores, const float* __restrict__ rw,
                        const float* __restrict__ rb, float* __restrict__ out) {
  const int b = threadIdx.x;  // 128
  float z = rb[0];
#pragma unroll
  for (int l = 0; l < L_; ++l) z = fmaf(scores[l * B_ + b], rw[l], z);
  out[b] = 1.f / (1.f + expf(-z));
}

// ---------------------------------------------------------------------------
extern "C" void kernel_launch(void* const* d_in, const int* in_sizes, int n_in,
                              void* d_out, int out_size, void* d_ws, size_t ws_size,
                              hipStream_t stream) {
  (void)in_sizes; (void)n_in; (void)out_size; (void)ws_size;
  const float* x    = (const float*)d_in[0];  // (128,3,224,224)
  const float* W    = (const float*)d_in[1];  // (4,150528,256)
  const float* mu   = (const float*)d_in[2];  // (4,1000,256)
  const float* Pr   = (const float*)d_in[3];  // (4,256,256)
  const float* stdv = (const float*)d_in[4];  // (3,)
  const float* rw   = (const float*)d_in[5];  // (1,4)
  const float* rb   = (const float*)d_in[6];  // (1,)
  float* out = (float*)d_out;                 // (128,)

  char* ws = (char*)d_ws;
  size_t off = 0;
  auto take = [&](size_t bytes) -> char* {
    char* p = ws + off;
    off += (bytes + 255) & ~(size_t)255;
    return p;
  };
  float* part  = (float*)take((size_t)L_ * NCH_ * B_ * D_ * 4);  // 25.7 MB
  float* outF  = (float*)take((size_t)L_ * B_ * D_ * 4);
  float* oP    = (float*)take((size_t)L_ * B_ * D_ * 4);
  float* out2  = (float*)take((size_t)L_ * B_ * D_ * 4);
  float* o2P   = (float*)take((size_t)L_ * B_ * D_ * 4);
  float* mPb   = (float*)take((size_t)L_ * C_ * D_ * 4);         // 4 MB
  float* mPm   = (float*)take((size_t)L_ * C_ * 4);
  float* xPx   = (float*)take((size_t)L_ * B_ * 4);
  float* x2Px2 = (float*)take((size_t)L_ * B_ * 4);
  unsigned short* zbf = (unsigned short*)take((size_t)L_ * B_ * D_ * 2);
  float* sc    = (float*)take((size_t)L_ * B_ * 4);

  const int nred = (L_ * B_ * D_) / 256;  // 512

  // pass 1: out = x @ W  (bf16 WMMA, split-K partials)
  k1_feat<<<dim3(L_ * NCH_ * 2), dim3(512), 0, stream>>>(x, W, part);
  k_reduce<<<dim3(nred), dim3(256), 0, stream>>>(part, nullptr, 1.f, outF);

  // quad-form prep + classify -> zfP = (out - mu[pred]) @ P  (bf16 for pass 2)
  k_quadprep<<<dim3(L_ * B_), dim3(256), 0, stream>>>(outF, Pr, B_, oP, xPx);
  k_quadprep<<<dim3(L_ * C_), dim3(256), 0, stream>>>(mu, Pr, C_, mPb, mPm);
  k_classify<<<dim3(L_ * B_), dim3(256), 0, stream>>>(oP, xPx, mu, mPb, mPm, 0, zbf, sc);

  // pass 2: fused gx = zfP@W^T -> sign -> corr = s@W  (one W stream)
  k3_fused<<<dim3(L_ * NCH_), dim3(512), 0, stream>>>(W, zbf, stdv, part);
  k_reduce<<<dim3(nred), dim3(256), 0, stream>>>(part, outF, -MAG_, out2);

  // scores = max_c quad(out2), then sigmoid regression head
  k_quadprep<<<dim3(L_ * B_), dim3(256), 0, stream>>>(out2, Pr, B_, o2P, x2Px2);
  k_classify<<<dim3(L_ * B_), dim3(256), 0, stream>>>(o2P, x2Px2, mu, mPb, mPm, 1, zbf, sc);
  k_final<<<dim3(1), dim3(B_), 0, stream>>>(sc, rw, rb, out);
}
